// GRUModel_90357521973683
// MI455X (gfx1250) — matrix operand order repaired and checked by reading, not compiled, not run on hardware
//
#include <hip/hip_runtime.h>

typedef _Float16 v16h __attribute__((ext_vector_type(16)));
typedef float    v8f  __attribute__((ext_vector_type(8)));
typedef _Float16 h8   __attribute__((ext_vector_type(8)));

#define B_TOT 4096
#define T_LEN 512
#define IN_D  28
#define HID   128
#define OUT_D 10
#define BT    16      // batch rows per workgroup
#define XPAD  40      // x-stage row stride in halfs (pads K 28->32, +8 skew)
#define HPAD  136     // h row stride in halfs (128 + 8 skew -> 4-bank rotation)

__device__ __forceinline__ v8f wmma_f16(v16h a, v16h b, v8f c) {
    return __builtin_amdgcn_wmma_f32_16x16x32_f16(false, a, false, b,
                                                  (short)0, c, false, false);
}

// Fast sigmoid / tanh: v_exp_f32 + v_rcp_f32 (no IEEE div chains)
__device__ __forceinline__ float fast_sigmoid(float x) {
    return __builtin_amdgcn_rcpf(1.f + __expf(-x));
}
__device__ __forceinline__ float fast_tanh(float x) {
    return 1.f - 2.f * __builtin_amdgcn_rcpf(1.f + __expf(2.f * x));
}

__global__ __launch_bounds__(256) void gru_fused_kernel(
    const float* __restrict__ x,     // [B, T, 28]
    const float* __restrict__ w_x,   // [3, 28, 128]
    const float* __restrict__ b_x,   // [3, 128]
    const float* __restrict__ w_h,   // [3, 128, 128]
    const float* __restrict__ b_h,   // [3, 128]
    const float* __restrict__ fc_w,  // [128, 10]
    const float* __restrict__ fc_b,  // [10]
    float* __restrict__ out)         // [B, 10]
{
    __shared__ __align__(16) _Float16 xs[BT * XPAD];   // staged x_t tile (f16)
    __shared__ __align__(16) _Float16 hs[BT * HPAD];   // h state (f16 shadow)
    __shared__ __align__(16) float    h32[BT * HID];   // final h (f32) for FC

    const int tid  = threadIdx.x;
    const int lane = tid & 31;
    const int wave = tid >> 5;            // 0..7 : hidden-column slice owner
    const int b0   = blockIdx.x * BT;     // batch-row base of this WG
    const int n0   = wave * 16;           // hidden column base of this wave
    const int nc   = n0 + (lane & 15);    // this lane's hidden column
    const int hi16 = lane >> 4;           // lane group (0: lanes 0-15, 1: 16-31)
    const int m    = lane & 15;           // A-fragment row
    const int koff = hi16 * 8;            // A-fragment K sub-offset

    // Zero h state and x staging (incl. K-pad columns 28..31, never rewritten)
    for (int i = tid; i < BT * HPAD; i += 256) hs[i] = (_Float16)0.f;
    for (int i = tid; i < BT * XPAD; i += 256) xs[i] = (_Float16)0.f;

    // ---- Load weight B-fragments into registers (resident for all 512 steps)
    // B layout (32x16 f16): lane holds column n = lane%16; elem i <-> K = 2*(i&8) + 8*hi16 + (i&7)
    v16h wxB[3];
#pragma unroll
    for (int g = 0; g < 3; ++g) {
        v16h f;
#pragma unroll
        for (int i = 0; i < 16; ++i) {
            int k = (i & 8) * 2 + hi16 * 8 + (i & 7);
            f[i] = (_Float16)((k < IN_D) ? w_x[(g * IN_D + k) * HID + nc] : 0.f);
        }
        wxB[g] = f;
    }
    v16h whB[3][4];
#pragma unroll
    for (int g = 0; g < 3; ++g)
#pragma unroll
        for (int c = 0; c < 4; ++c) {
            v16h f;
#pragma unroll
            for (int i = 0; i < 16; ++i) {
                int k = c * 32 + (i & 8) * 2 + hi16 * 8 + (i & 7);
                f[i] = (_Float16)w_h[(g * HID + k) * HID + nc];
            }
            whB[g][c] = f;
        }

    // Biases folded into accumulator init (per output column nc)
    const float bias_r  = b_x[0 * HID + nc] + b_h[0 * HID + nc];
    const float bias_z  = b_x[1 * HID + nc] + b_h[1 * HID + nc];
    const float bias_nx = b_x[2 * HID + nc];
    const float bias_nh = b_h[2 * HID + nc];

    // f32 register-resident h in WMMA C layout: elem j -> row (j + 8*hi16), col nc
    v8f hreg = {};

    // Per-thread x-staging / prefetch coordinates (16*28 = 448 elems, 256 thr)
    const int e0_m = tid / IN_D, e0_i = tid - e0_m * IN_D;
    const int e1 = tid + 256;
    const int e1_m = e1 / IN_D, e1_i = e1 - e1_m * IN_D;
    const float* xrow0 = x + ((size_t)(b0 + e0_m) * T_LEN) * IN_D + e0_i;
    const float* xrow1 = x + ((size_t)(b0 + e1_m) * T_LEN) * IN_D + e1_i;

    __syncthreads();   // hs/xs zeroing visible before the time loop

    for (int t = 0; t < T_LEN; ++t) {
        // Stage x_t tile -> LDS (coalesced, f32 -> f16); 448 elems, <=2 per thr
        xs[e0_m * XPAD + e0_i] = (_Float16)xrow0[(size_t)t * IN_D];
        if (e1 < BT * IN_D)
            xs[e1_m * XPAD + e1_i] = (_Float16)xrow1[(size_t)t * IN_D];
        // Prefetch next step's x tile into cache
        if (t + 1 < T_LEN)
            __builtin_prefetch(&xrow0[(size_t)(t + 1) * IN_D], 0, 0);
        __syncthreads();   // xs ready; hs writes from prev step visible

        // ---- Hoist ALL A-fragments (x + 4 h chunks) so WMMAs issue densely
        v16h xa, ha[4];
        {
            h8 lo = *(const h8*)&xs[m * XPAD + koff];
            h8 hi = *(const h8*)&xs[m * XPAD + 16 + koff];
#pragma unroll
            for (int i = 0; i < 8; ++i) { xa[i] = lo[i]; xa[8 + i] = hi[i]; }
        }
#pragma unroll
        for (int c = 0; c < 4; ++c) {
            h8 lo = *(const h8*)&hs[m * HPAD + c * 32 + koff];
            h8 hi = *(const h8*)&hs[m * HPAD + c * 32 + 16 + koff];
#pragma unroll
            for (int i = 0; i < 8; ++i) { ha[c][i] = lo[i]; ha[c][8 + i] = hi[i]; }
        }

        v8f accR, accZ, accNX, accNH;
#pragma unroll
        for (int j = 0; j < 8; ++j) {
            accR[j] = bias_r; accZ[j] = bias_z;
            accNX[j] = bias_nx; accNH[j] = bias_nh;
        }

        // Input projections (K = 32, one WMMA each)
        accR  = wmma_f16(xa, wxB[0], accR);
        accZ  = wmma_f16(xa, wxB[1], accZ);
        accNX = wmma_f16(xa, wxB[2], accNX);
        // Hidden projections (K = 128 -> 4 chunks of 32)
#pragma unroll
        for (int c = 0; c < 4; ++c) {
            accR  = wmma_f16(ha[c], whB[0][c], accR);
            accZ  = wmma_f16(ha[c], whB[1][c], accZ);
            accNH = wmma_f16(ha[c], whB[2][c], accNH);
        }

        // Gate combine (f32, wave-local in C layout) — fully unrolled
        v8f hnew;
#pragma unroll
        for (int j = 0; j < 8; ++j) {
            float r   = fast_sigmoid(accR[j]);
            float z   = fast_sigmoid(accZ[j]);
            float n   = fast_tanh(accNX[j] + r * accNH[j]);
            hnew[j]   = n + z * (hreg[j] - n);   // (1-z)*n + z*h
        }
        hreg = hnew;

        __syncthreads();   // all A-fragment reads of hs complete
        // Publish h_next (f16 shadow) for next step's A-operand
#pragma unroll
        for (int j = 0; j < 8; ++j)
            hs[(j + 8 * hi16) * HPAD + nc] = (_Float16)hreg[j];
        // visibility guaranteed by next iteration's post-stage barrier
    }

    // ---- Final FC: out = h_last @ fc_w + fc_b  (16 x 10 per WG, f32 path)
#pragma unroll
    for (int j = 0; j < 8; ++j)
        h32[(j + 8 * hi16) * HID + nc] = hreg[j];
    __syncthreads();

    for (int e = tid; e < BT * OUT_D; e += 256) {
        int mm = e / OUT_D, o = e - mm * OUT_D;
        float acc = fc_b[o];
#pragma unroll 4
        for (int k = 0; k < HID; ++k)
            acc += h32[mm * HID + k] * fc_w[k * OUT_D + o];
        out[(size_t)(b0 + mm) * OUT_D + o] = acc;
    }
}

extern "C" void kernel_launch(void* const* d_in, const int* in_sizes, int n_in,
                              void* d_out, int out_size, void* d_ws, size_t ws_size,
                              hipStream_t stream) {
    const float* x    = (const float*)d_in[0];
    const float* w_x  = (const float*)d_in[1];
    const float* b_x  = (const float*)d_in[2];
    const float* w_h  = (const float*)d_in[3];
    const float* b_h  = (const float*)d_in[4];
    const float* fc_w = (const float*)d_in[5];
    const float* fc_b = (const float*)d_in[6];
    float* out = (float*)d_out;

    dim3 grid(B_TOT / BT);   // 256 workgroups
    dim3 block(256);         // 8 wave32 per WG
    gru_fused_kernel<<<grid, block, 0, stream>>>(x, w_x, b_x, w_h, b_h,
                                                 fc_w, fc_b, out);
}
